// FNO2d_48275432407351
// MI455X (gfx1250) — matrix operand order repaired and checked by reading, not compiled, not run on hardware
//
#include <hip/hip_runtime.h>
#include <math.h>

// ---------------------------------------------------------------------------
// FNO2d for MI455X (gfx1250): bf16 WMMA GEMMs for all channel mixing and the
// partial-DFT spectral path (only 16x16 modes survive, so rfft2/irfft2
// collapse into small DFT GEMMs against cos/sin tables). A-tiles are staged
// into LDS by the Tensor Data Mover (TENSORcnt) where possible.
//
// Shapes: B=4, Cin=3, H=W=256, width=64, m1=m2=16, L=4, fc1=128, out=1.
// Activation layout: channels-last bf16  hb[(b*H+h)*W+w][64].
// ---------------------------------------------------------------------------

typedef __attribute__((ext_vector_type(16))) __bf16 v16bf;
typedef __attribute__((ext_vector_type(8)))  float  v8f;
typedef __attribute__((ext_vector_type(4)))  unsigned int u32x4;
typedef __attribute__((ext_vector_type(8)))  int          i32x8;
typedef __attribute__((ext_vector_type(4)))  int          i32x4;
struct Frag8 { unsigned int w[8]; };

#if defined(__has_builtin)
#if __has_builtin(__builtin_amdgcn_tensor_load_to_lds) && __has_builtin(__builtin_amdgcn_s_wait_tensorcnt)
#define HAVE_TDM 1
#else
#define HAVE_TDM 0
#endif
#else
#define HAVE_TDM 0
#endif

__device__ __forceinline__ unsigned short f2bf(float x) {
    unsigned int u = __builtin_bit_cast(unsigned int, x);
    unsigned int r = u + 0x7FFFu + ((u >> 16) & 1u);   // round-to-nearest-even
    return (unsigned short)(r >> 16);
}
__device__ __forceinline__ unsigned int pack2bf(float lo, float hi) {
    return (unsigned int)f2bf(lo) | ((unsigned int)f2bf(hi) << 16);
}
__device__ __forceinline__ float bf2f(unsigned short h) {
    unsigned int u = ((unsigned int)h) << 16;
    return __builtin_bit_cast(float, u);
}
__device__ __forceinline__ float gelu_exact(float x) {
    return 0.5f * x * (1.0f + erff(x * 0.70710678118654752f));
}

#if HAVE_TDM
// Issue one TDM 2D tile load: 64 rows x 32 bf16, row stride = rs elements,
// into LDS at lds_off with 16B padding after each 64B row (LDS row = 80B).
// D# layout per CDNA5 ISA 8.3/8.4 (group0 128b, group1 256b; groups 2/3 unused).
__device__ __forceinline__ void tdm_load_tile_a(const unsigned short* gptr,
                                                unsigned lds_off, unsigned rs)
{
    unsigned long long ga = (unsigned long long)(uintptr_t)gptr;
    u32x4 g0 = {
        1u,                                               // count=1 (valid), user mode
        lds_off,                                          // lds_addr (bytes)
        (unsigned)(ga & 0xFFFFFFFFu),                     // global_addr[31:0]
        (unsigned)((ga >> 32) & 0x01FFFFFFu) | (2u << 30) // global_addr[56:32] | type=2
    };
    i32x8 g1 = {
        (1 << 16) | (1 << 20) | (3 << 22) | (3 << 25),    // data_size=2B, pad_en, 16dw interval, +4dw pad
        (int)(32u << 16),                                 // tensor_dim0 = 32  (bits 79:48)
        (int)(64u << 16),                                 // tensor_dim1 = 64  (bits 111:80)
        (int)(32u << 16),                                 // tile_dim0  = 32  (bits 127:112)
        64,                                               // tile_dim1 = 64, tile_dim2 = 0
        (int)rs,                                          // tensor_dim0_stride (elements)
        0, 0                                              // stride0 hi / dim1_stride (unused, 2D)
    };
    i32x4 z4 = {0, 0, 0, 0};
#if defined(__clang_major__) && (__clang_major__ >= 23)
    i32x8 z8 = {0, 0, 0, 0, 0, 0, 0, 0};
    __builtin_amdgcn_tensor_load_to_lds(g0, g1, z4, z4, z8, 0);
#else
    __builtin_amdgcn_tensor_load_to_lds(g0, g1, z4, z4, 0);
#endif
}
#endif

// ---------------------------------------------------------------------------
// Strided, batched bf16 GEMM:  C[batch] (+)= A[batch] * B[batch]
//   A: bf16 at A + batch*a_bs + m*a_rs + k*a_cs
//   B: bf16 row-major K x N (ldb) + batch*b_bs
//   C: f32 or bf16 at Cv + batch*c_bs + m*c_rs + n*c_cs
// Block: 256 threads = 8 wave32s; block tile 64(M) x 64(N); K step 32.
// Wave w: wm=w&3 (16-row slab), wn=w>>2 (32-col slab) -> 2 WMMAs sharing the
// A fragment. Requires M%64==0, N%32==0, K%32==0 (true for all calls here).
// AMODE 0: a_cs==1 (k-contiguous rows) -> TDM tile load (or b128 fallback)
// AMODE 1: a_rs==1 (m-contiguous cols) -> b128 load + transpose scatter
// FLAGS (compile time): 1 = accumulate into f32 C, 2 = GeLU, 4 = store bf16
// ---------------------------------------------------------------------------
template <int AMODE, int FLAGS, bool HAS_BIAS>
__global__ __launch_bounds__(256) void gemm_wmma_bf16(
    const unsigned short* __restrict__ A, long a_rs, long a_cs, long a_bs,
    const unsigned short* __restrict__ B, int ldb, long b_bs,
    void* __restrict__ Cv, long c_rs, long c_cs, long c_bs,
    int M, int N, int K, const float* __restrict__ bias)
{
    // row stride 20 dwords (80 B): 16B-aligned vector stores, bank de-phasing
    __shared__ __align__(16) unsigned int As[64 * 20];
    __shared__ __align__(16) unsigned int Bs[64 * 20];
    unsigned short* As16 = (unsigned short*)As;
    unsigned short* Bs16 = (unsigned short*)Bs;

    const int tid   = threadIdx.x;
    const int lane  = tid & 31;
    const int wave  = tid >> 5;
    const int wm    = wave & 3;
    const int wn    = wave >> 2;           // 0 | 1
    const int lhalf = lane >> 4;           // 0 | 1
    const int l16   = lane & 15;
    const int wave_u = __builtin_amdgcn_readfirstlane(wave);   // provably uniform

    const int m0 = blockIdx.y * 64;
    const int n0 = blockIdx.x * 64;
    const long batch = blockIdx.z;
    const unsigned short* Ab = A + batch * a_bs;
    const unsigned short* Bb = B + batch * b_bs;
    const bool active = (n0 + wn * 32) < N;      // wave-uniform (EXEC stays full)

    v8f acc0 = {0.f,0.f,0.f,0.f,0.f,0.f,0.f,0.f};
    v8f acc1 = acc0;

    for (int k0 = 0; k0 < K; k0 += 32) {
        // ---- stage A tile (64 x 32) ----
        if (AMODE == 0) {
#if HAVE_TDM
            if (wave_u == 0)       // one TDM descriptor moves the whole tile
                tdm_load_tile_a(Ab + (long)m0 * a_rs + k0,
                                (unsigned)(uintptr_t)As, (unsigned)a_rs);
#else
            const int m = tid >> 2, koff = (tid & 3) * 8;
            uint4 v = *(const uint4*)(Ab + (long)(m0 + m) * a_rs + (k0 + koff));
            *(uint4*)(As16 + m * 40 + koff) = v;
#endif
        } else {
            const int k = tid >> 3, moff = (tid & 7) * 8;
            uint4 v = *(const uint4*)(Ab + (long)(k0 + k) * a_cs + (m0 + moff));
            const unsigned short* e = (const unsigned short*)&v;
            #pragma unroll
            for (int j = 0; j < 8; ++j) As16[(moff + j) * 40 + k] = e[j];
        }
        // ---- stage B tile (32 x 64), transposed to [n][k] (all waves) ----
        {
            const int k = tid >> 3, noff = (tid & 7) * 8;
            uint4 v = {0u, 0u, 0u, 0u};
            if (n0 + noff < N)
                v = *(const uint4*)(Bb + (long)(k0 + k) * ldb + (n0 + noff));
            const unsigned short* e = (const unsigned short*)&v;
            #pragma unroll
            for (int j = 0; j < 8; ++j) Bs16[(noff + j) * 40 + k] = e[j];
        }
        if (k0 + 32 < K) {   // warm GL2 for the next A chunk
            if (AMODE == 0)
                __builtin_prefetch(Ab + (long)(m0 + (tid >> 2)) * a_rs + (k0 + 32), 0, 1);
            else
                __builtin_prefetch(Ab + (long)(k0 + 32 + (tid >> 3)) * a_cs + m0, 0, 1);
        }
#if HAVE_TDM
        if (AMODE == 0 && wave_u == 0)
            __builtin_amdgcn_s_wait_tensorcnt(0);
#endif
        __syncthreads();

        if (active) {
            // fragments per the documented 16-bit A/B wave32 VGPR layouts
            Frag8 fa, fb0, fb1;
            #pragma unroll
            for (int r = 0; r < 8; ++r) {
                int kp = ((r >> 2) << 3) + (lhalf << 2) + (r & 3);
                fa.w[r]  = As[(wm * 16 + l16) * 20 + kp];
                fb0.w[r] = Bs[(wn * 32      + l16) * 20 + kp];
                fb1.w[r] = Bs[(wn * 32 + 16 + l16) * 20 + kp];
            }
            v16bf av = __builtin_bit_cast(v16bf, fa);
            acc0 = __builtin_amdgcn_wmma_f32_16x16x32_bf16(
                       false, av, false, __builtin_bit_cast(v16bf, fb0), (short)0, acc0, false, false);
            acc1 = __builtin_amdgcn_wmma_f32_16x16x32_bf16(
                       false, av, false, __builtin_bit_cast(v16bf, fb1), (short)0, acc1, false, false);
        }
        __syncthreads();
    }

    if (!active) return;
    // C layout: lane = N column, VGPR r -> M = r + 8*lhalf
    const int row  = m0 + wm * 16 + lhalf * 8;
    const int col0 = n0 + wn * 32 + l16;
    const float b0 = HAS_BIAS ? bias[col0] : 0.f;
    const float b1 = HAS_BIAS ? bias[col0 + 16] : 0.f;
    #pragma unroll
    for (int r = 0; r < 8; ++r) {
        long o0 = batch * c_bs + (long)(row + r) * c_rs + (long)col0 * c_cs;
        long o1 = o0 + 16L * c_cs;
        float v0 = acc0[r] + b0, v1 = acc1[r] + b1;
        if (FLAGS & 1) { v0 += ((const float*)Cv)[o0]; v1 += ((const float*)Cv)[o1]; }
        if (FLAGS & 2) { v0 = gelu_exact(v0); v1 = gelu_exact(v1); }
        if (FLAGS & 4) { ((unsigned short*)Cv)[o0] = f2bf(v0); ((unsigned short*)Cv)[o1] = f2bf(v1); }
        else           { ((float*)Cv)[o0] = v0; ((float*)Cv)[o1] = v1; }
    }
}

// ---------------------------------------------------------------------------
// Tables: forward-W DFT (bf16, 256x32), inverse-W DFT with Hermitian fold and
// 1/(H*W) scale (bf16, 32x256), f32 cos/sin for the H axis (256x16 each).
// ---------------------------------------------------------------------------
__global__ void init_tables_kernel(unsigned short* __restrict__ FW,
                                   unsigned short* __restrict__ EW2,
                                   float* __restrict__ cosH,
                                   float* __restrict__ sinH)
{
    const int w = threadIdx.x;                  // 0..255
    const float twopi = 6.28318530717958647692f;
    for (int k = 0; k < 16; ++k) {
        float th = twopi * (float)(k * w) / 256.0f;
        float s, c;
        __sincosf(th, &s, &c);
        FW[w * 32 + 2 * k]     = f2bf(c);       // X[k] = sum_w x (cos - i sin)
        FW[w * 32 + 2 * k + 1] = f2bf(-s);
        cosH[w * 16 + k] = c;
        sinH[w * 16 + k] = s;
        float sc = ((k == 0) ? 1.0f : 2.0f) / 65536.0f;
        EW2[(2 * k) * 256 + w]     = f2bf(sc * c);   // Re{c_k (Tr+iTi)(cos+isin)}
        EW2[(2 * k + 1) * 256 + w] = f2bf(-sc * s);
    }
}

// Cast/transpose weights to bf16 GEMM-B layout: Wbf[l][c][o], fc1bf[c][o].
__global__ void prep_weights_kernel(const float* __restrict__ w_w,    // (L,64,64)
                                    const float* __restrict__ fc1_w,  // (128,64)
                                    unsigned short* __restrict__ Wbf,
                                    unsigned short* __restrict__ fc1bf)
{
    int idx = blockIdx.x * 256 + threadIdx.x;
    if (idx < 4 * 64 * 64) {
        int l = idx >> 12, rem = idx & 4095;
        int o = rem >> 6, c = rem & 63;
        Wbf[l * 4096 + c * 64 + o] = f2bf(w_w[l * 4096 + o * 64 + c]);
    } else if (idx < 4 * 64 * 64 + 128 * 64) {
        int j = idx - 4 * 64 * 64;
        int o = j >> 6, c = j & 63;
        fc1bf[c * 128 + o] = f2bf(fc1_w[o * 64 + c]);
    }
}

// fc0: 3->64 pointwise lift; channels-first f32 in -> channels-last bf16 out.
__global__ __launch_bounds__(256) void fc0_kernel(
    const float* __restrict__ x,      // (4,3,65536)
    const float* __restrict__ w,      // (64,3)
    const float* __restrict__ b,      // (64,)
    unsigned short* __restrict__ hb)  // (262144,64)
{
    int tid = blockIdx.x * 256 + threadIdx.x;      // (b,hw)
    int bi = tid >> 16, hw = tid & 65535;
    float x0 = x[(bi * 3 + 0) * 65536 + hw];
    float x1 = x[(bi * 3 + 1) * 65536 + hw];
    float x2 = x[(bi * 3 + 2) * 65536 + hw];
    uint4* dst = (uint4*)(hb + (long)tid * 64);
    #pragma unroll
    for (int g = 0; g < 8; ++g) {
        float v[8];
        #pragma unroll
        for (int j = 0; j < 8; ++j) {
            int o = g * 8 + j;
            v[j] = b[o] + w[o * 3] * x0 + w[o * 3 + 1] * x1 + w[o * 3 + 2] * x2;
        }
        uint4 q;
        q.x = pack2bf(v[0], v[1]); q.y = pack2bf(v[2], v[3]);
        q.z = pack2bf(v[4], v[5]); q.w = pack2bf(v[6], v[7]);
        dst[g] = q;
    }
}

// Forward DFT along H: Xm[b][c][kx][ky][2] = sum_h T1 * (cos - i sin).
__global__ __launch_bounds__(256) void dft_h_kernel(
    const float* __restrict__ T1,      // [(b*256+h)][c][32]
    float* __restrict__ Xm,
    const float* __restrict__ cosH, const float* __restrict__ sinH)
{
    int idx = blockIdx.x * 256 + threadIdx.x;      // 65536 = b*c*kx*ky
    int ky = idx & 15, kx = (idx >> 4) & 15, c = (idx >> 8) & 63, bi = idx >> 14;
    float xr = 0.f, xi = 0.f;
    for (int h = 0; h < 256; ++h) {
        long t = (((long)(bi * 256 + h)) * 64 + c) * 32 + 2 * ky;
        float Tr = T1[t], Ti = T1[t + 1];
        float ch = cosH[h * 16 + kx], sh = sinH[h * 16 + kx];
        xr += Tr * ch + Ti * sh;
        xi += Ti * ch - Tr * sh;
    }
    long o = ((((long)bi * 64 + c) * 16 + kx) * 16 + ky) * 2;
    Xm[o] = xr; Xm[o + 1] = xi;
}

// Complex mode mixing: blk[b][o][kx][ky] = sum_i Xm[b][i] * (wr + i wi)[i][o].
__global__ __launch_bounds__(256) void modemix_kernel(
    const float* __restrict__ Xm,
    const float* __restrict__ wr, const float* __restrict__ wi,  // (64,64,16,16)
    float* __restrict__ blk)
{
    int idx = blockIdx.x * 256 + threadIdx.x;      // 65536 = b*o*kx*ky
    int ky = idx & 15, kx = (idx >> 4) & 15, o = (idx >> 8) & 63, bi = idx >> 14;
    float br = 0.f, bim = 0.f;
    int mo = kx * 16 + ky;
    for (int i = 0; i < 64; ++i) {
        long xo = ((((long)bi * 64 + i) * 256) + mo) * 2;
        float Xr = Xm[xo], Xi = Xm[xo + 1];
        float Wr = wr[((long)i * 64 + o) * 256 + mo];
        float Wi = wi[((long)i * 64 + o) * 256 + mo];
        br  += Xr * Wr - Xi * Wi;
        bim += Xr * Wi + Xi * Wr;
    }
    long bo = ((((long)bi * 64 + o) * 256) + mo) * 2;
    blk[bo] = br; blk[bo + 1] = bim;
}

// Inverse DFT along H: T2[(b*64+o)*256+h][2*ky{re,im}] bf16 (GEMM A operand).
__global__ __launch_bounds__(256) void idft_h_kernel(
    const float* __restrict__ blk,
    unsigned short* __restrict__ T2,
    const float* __restrict__ cosH, const float* __restrict__ sinH)
{
    int idx = blockIdx.x * 256 + threadIdx.x;      // 65536 = (b,o,h)
    int h = idx & 255, o = (idx >> 8) & 63, bi = idx >> 14;
    float ar[16], ai[16];
    #pragma unroll
    for (int k = 0; k < 16; ++k) { ar[k] = 0.f; ai[k] = 0.f; }
    for (int kx = 0; kx < 16; ++kx) {
        float ch = cosH[h * 16 + kx], sh = sinH[h * 16 + kx];
        long base = ((((long)bi * 64 + o) * 16 + kx) * 16) * 2;
        #pragma unroll
        for (int ky = 0; ky < 16; ++ky) {
            float Br = blk[base + 2 * ky], Bi = blk[base + 2 * ky + 1];
            ar[ky] += Br * ch - Bi * sh;       // *(cos + i sin)
            ai[ky] += Br * sh + Bi * ch;
        }
    }
    uint4* dst = (uint4*)(T2 + (long)idx * 32);
    #pragma unroll
    for (int g = 0; g < 4; ++g) {
        uint4 q;
        q.x = pack2bf(ar[g*4+0], ai[g*4+0]); q.y = pack2bf(ar[g*4+1], ai[g*4+1]);
        q.z = pack2bf(ar[g*4+2], ai[g*4+2]); q.w = pack2bf(ar[g*4+3], ai[g*4+3]);
        dst[g] = q;
    }
}

// acc[b][o][h][w] (conv + spectral) -> +bias, GeLU, back to channels-last bf16.
__global__ __launch_bounds__(256) void combine_kernel(
    const float* __restrict__ accb,   // (4,64,65536)
    const float* __restrict__ wb,     // (64,)
    unsigned short* __restrict__ hb)  // (262144,64)
{
    int tid = blockIdx.x * 256 + threadIdx.x;      // (b,hw)
    int bi = tid >> 16, hw = tid & 65535;
    uint4* dst = (uint4*)(hb + (long)tid * 64);
    #pragma unroll
    for (int g = 0; g < 8; ++g) {
        float v[8];
        #pragma unroll
        for (int j = 0; j < 8; ++j) {
            int o = g * 8 + j;
            v[j] = gelu_exact(accb[((long)(bi * 64 + o)) * 65536 + hw] + wb[o]);
        }
        uint4 q;
        q.x = pack2bf(v[0], v[1]); q.y = pack2bf(v[2], v[3]);
        q.z = pack2bf(v[4], v[5]); q.w = pack2bf(v[6], v[7]);
        dst[g] = q;
    }
}

// fc2: 128 -> 1 projection (N=1 -> dot product per pixel).
__global__ __launch_bounds__(256) void fc2_kernel(
    const unsigned short* __restrict__ g1,   // (262144,128) bf16
    const float* __restrict__ w,             // (1,128)
    const float* __restrict__ b,             // (1,)
    float* __restrict__ out)                 // (262144,)
{
    int tid = blockIdx.x * 256 + threadIdx.x;
    const uint4* src = (const uint4*)(g1 + (long)tid * 128);
    float acc = b[0];
    #pragma unroll
    for (int g = 0; g < 16; ++g) {
        uint4 q = src[g];
        unsigned int u[4] = {q.x, q.y, q.z, q.w};
        #pragma unroll
        for (int j = 0; j < 4; ++j) {
            acc += bf2f((unsigned short)(u[j] & 0xFFFF)) * w[g * 8 + 2 * j];
            acc += bf2f((unsigned short)(u[j] >> 16))    * w[g * 8 + 2 * j + 1];
        }
    }
    out[tid] = acc;
}

// ---------------------------------------------------------------------------
extern "C" void kernel_launch(void* const* d_in, const int* in_sizes, int n_in,
                              void* d_out, int out_size, void* d_ws, size_t ws_size,
                              hipStream_t stream) {
    const float* x       = (const float*)d_in[0];
    const float* fc0_w   = (const float*)d_in[1];
    const float* fc0_b   = (const float*)d_in[2];
    const float* spec_wr = (const float*)d_in[3];   // (4,64,64,16,16)
    const float* spec_wi = (const float*)d_in[4];
    const float* w_w     = (const float*)d_in[5];   // (4,64,64)
    const float* w_b     = (const float*)d_in[6];   // (4,64)
    const float* fc1_w   = (const float*)d_in[7];   // (128,64)
    const float* fc1_b   = (const float*)d_in[8];
    const float* fc2_w   = (const float*)d_in[9];
    const float* fc2_b   = (const float*)d_in[10];
    float* out = (float*)d_out;

    // workspace carve-up (byte offsets, 256B aligned)
    char* ws = (char*)d_ws;
    unsigned short* hb    = (unsigned short*)(ws);                       // 32 MB  bf16 act
    float*          accb  = (float*)(ws + 33554432);                     // 64 MB  conv+spec f32
    float*          T1    = (float*)(ws + 100663296);                    // 8.4 MB
    unsigned short* T2    = (unsigned short*)(ws + 109051904);           // 4.2 MB
    float*          Xm    = (float*)(ws + 113246208);                    // 0.5 MB
    float*          blkb  = (float*)(ws + 113770496);                    // 0.5 MB
    unsigned short* g1    = (unsigned short*)(ws + 114294784);           // 67 MB
    unsigned short* FW    = (unsigned short*)(ws + 181403648);           // 16 KB
    unsigned short* EW2   = (unsigned short*)(ws + 181420032);           // 16 KB
    float*          cosH  = (float*)(ws + 181436416);                    // 16 KB
    float*          sinH  = (float*)(ws + 181452800);                    // 16 KB
    unsigned short* Wbf   = (unsigned short*)(ws + 181469184);           // 32 KB
    unsigned short* fc1bf = (unsigned short*)(ws + 181501952);           // 16 KB

    init_tables_kernel<<<1, 256, 0, stream>>>(FW, EW2, cosH, sinH);
    prep_weights_kernel<<<96, 256, 0, stream>>>(w_w, fc1_w, Wbf, fc1bf);
    fc0_kernel<<<1024, 256, 0, stream>>>(x, fc0_w, fc0_b, hb);

    for (int l = 0; l < 4; ++l) {
        // conv1x1 64->64: batch over b; C written channel-first into accb
        gemm_wmma_bf16<0, 0, false><<<dim3(1, 1024, 4), 256, 0, stream>>>(
            hb, 64L, 1L, 65536L * 64L,
            Wbf + l * 4096, 64, 0L,
            accb, 1L, 65536L, 64L * 65536L,
            65536, 64, 64, nullptr);

        // forward partial DFT along W: per-(b,h) GEMM (64c x 256w)*(256w x 32)
        gemm_wmma_bf16<1, 0, false><<<dim3(1, 1, 1024), 256, 0, stream>>>(
            hb, 1L, 64L, 16384L,
            FW, 32, 0L,
            T1, 32L, 1L, 2048L,
            64, 32, 256, nullptr);

        dft_h_kernel<<<256, 256, 0, stream>>>(T1, Xm, cosH, sinH);
        modemix_kernel<<<256, 256, 0, stream>>>(
            Xm, spec_wr + (long)l * 1048576, spec_wi + (long)l * 1048576, blkb);
        idft_h_kernel<<<256, 256, 0, stream>>>(blkb, T2, cosH, sinH);

        // inverse DFT along W, accumulating onto the conv result in accb
        gemm_wmma_bf16<0, 1, false><<<dim3(4, 1024, 1), 256, 0, stream>>>(
            T2, 32L, 1L, 0L,
            EW2, 256, 0L,
            accb, 256L, 1L, 0L,
            65536, 256, 32, nullptr);

        combine_kernel<<<1024, 256, 0, stream>>>(accb, w_b + l * 64, hb);
    }

    // fc1 64->128 with bias + GeLU, bf16 output
    gemm_wmma_bf16<0, 6, true><<<dim3(2, 4096, 1), 256, 0, stream>>>(
        hb, 64L, 1L, 0L,
        fc1bf, 128, 0L,
        g1, 128L, 1L, 0L,
        262144, 128, 64, fc1_b);

    fc2_kernel<<<1024, 256, 0, stream>>>(g1, fc2_w, fc2_b, out);
}
